// G_Spatial_Adversarial_50225347559669
// MI455X (gfx1250) — compile-verified, matrix operand order checked by reading, not run on hardware
//
#include <hip/hip_runtime.h>
#include <hip/hip_bf16.h>

#define B_   256
#define T_   32
#define E_   512
#define H_   512
#define V_   3000
#define G4H  2048   // 4*H
#define D2H  1024   // 2*H
#define WOUT_PAD 3072   // V padded to 24*128 column-blocks
#define SPLITK_H 4      // split-K factor for the sequential hidden GEMM

typedef __attribute__((ext_vector_type(16))) __bf16 v16bf;
typedef __attribute__((ext_vector_type(8)))  __bf16 v8bf;
typedef __attribute__((ext_vector_type(8)))  float  v8f;

__device__ __forceinline__ float sigmoidf_(float x) { return 1.f / (1.f + __expf(-x)); }

// ---------------------------------------------------------------------------
// Generic bf16 WMMA GEMM:  C[M x N] = A[M x K] * W[N x K]^T  (+ optional bias)
// A, W row-major bf16 with row stride ldk. Wave tile = 16*MT x 16*NT; block =
// 4 waves (2x2) -> block tile 32*MT x 32*NT. Grid: x = N/(32*NT), y = M/(32*MT),
// z = split-K chunk (each chunk is kChunk wide; partials go to C + z*cChunkStride).
// A/B fragment layout per CDNA5 ISA 16-bit operand tables:
//   lane<16 : M(or N)=lane,    K in {kb+0..7,  kb+16..23}
//   lane>=16: M(or N)=lane-16, K in {kb+8..15, kb+24..31}
// ---------------------------------------------------------------------------
template <int MT, int NT>
__global__ __launch_bounds__(128)
void k_gemm(const __bf16* __restrict__ A, const __bf16* __restrict__ W,
            float* __restrict__ C, int kChunk, int ldk, int ldc,
            int Mstore, int Nstore, const float* __restrict__ bias,
            size_t cChunkStride)
{
    const int lane = threadIdx.x & 31;
    const int wid  = threadIdx.x >> 5;                 // 0..3
    const int wm   = wid & 1;
    const int wn   = wid >> 1;
    const int m0   = blockIdx.y * (32 * MT) + wm * (16 * MT);
    const int n0   = blockIdx.x * (32 * NT) + wn * (16 * NT);
    const int kbase = blockIdx.z * kChunk;

    const int kh  = lane >> 4;                         // which K-half this lane holds
    const int l16 = lane & 15;

    v8f acc[MT][NT];
    #pragma unroll
    for (int i = 0; i < MT; ++i)
        #pragma unroll
        for (int j = 0; j < NT; ++j)
            #pragma unroll
            for (int r = 0; r < 8; ++r) acc[i][j][r] = 0.f;

    const __bf16* arow[MT];
    const __bf16* wrow[NT];
    #pragma unroll
    for (int i = 0; i < MT; ++i) arow[i] = A + (size_t)(m0 + 16 * i + l16) * ldk;
    #pragma unroll
    for (int j = 0; j < NT; ++j) wrow[j] = W + (size_t)(n0 + 16 * j + l16) * ldk;

    for (int kk = 0; kk < kChunk; kk += 32) {
        const int o0 = kbase + kk + kh * 8;        // first 8 bf16 (16B) this lane holds
        const int o1 = kbase + kk + 16 + kh * 8;   // second 8 bf16

        v16bf afrag[MT];
        #pragma unroll
        for (int i = 0; i < MT; ++i) {
            v8bf lo = *(const v8bf*)(arow[i] + o0);
            v8bf hi = *(const v8bf*)(arow[i] + o1);
            #pragma unroll
            for (int e = 0; e < 8; ++e) { afrag[i][e] = lo[e]; afrag[i][8 + e] = hi[e]; }
        }

        #pragma unroll
        for (int j = 0; j < NT; ++j) {
            v16bf bb;
            v8bf lo = *(const v8bf*)(wrow[j] + o0);
            v8bf hi = *(const v8bf*)(wrow[j] + o1);
            #pragma unroll
            for (int e = 0; e < 8; ++e) { bb[e] = lo[e]; bb[8 + e] = hi[e]; }

            #pragma unroll
            for (int i = 0; i < MT; ++i)
                acc[i][j] = __builtin_amdgcn_wmma_f32_16x16x32_bf16(
                    false, afrag[i], false, bb, (short)0, acc[i][j], false, false);
        }
    }

    float* Cz = C + (size_t)blockIdx.z * cChunkStride;
    // C/D layout: VGPR r -> row = tile_m + 8*kh + r ; col = tile_n + l16
    #pragma unroll
    for (int i = 0; i < MT; ++i)
        #pragma unroll
        for (int j = 0; j < NT; ++j)
            #pragma unroll
            for (int r = 0; r < 8; ++r) {
                int row = m0 + 16 * i + 8 * kh + r;
                int col = n0 + 16 * j + l16;
                if (row < Mstore && col < Nstore) {
                    float bv = bias ? bias[col] : 0.f;
                    Cz[(size_t)row * ldc + col] = acc[i][j][r] + bv;
                }
            }
}

// ---------------------------------------------------------------------------
// Build X[T*B, E] in bf16: t=0 -> concat(image, text), t>=1 -> embed[answer[:,t-1]]
// ---------------------------------------------------------------------------
__global__ void k_build_x(const float* __restrict__ img, const float* __restrict__ txt,
                          const float* __restrict__ emb, const int* __restrict__ answer,
                          __bf16* __restrict__ X)
{
    const size_t total = (size_t)T_ * B_ * E_;
    for (size_t i = blockIdx.x * (size_t)blockDim.x + threadIdx.x; i < total;
         i += (size_t)gridDim.x * blockDim.x) {
        int t   = (int)(i >> 17);              // / (B_*E_) = 131072
        int rem = (int)(i & 131071);
        int b   = rem >> 9;                    // / E_
        int e   = rem & 511;
        float v;
        if (t == 0)
            v = (e < E_ / 2) ? img[b * (E_ / 2) + e] : txt[b * (E_ / 2) + (e - E_ / 2)];
        else {
            int tok = answer[b * T_ + (t - 1)];
            v = emb[(size_t)tok * E_ + e];
        }
        X[i] = (__bf16)v;
    }
}

__global__ void k_cvt(const float* __restrict__ s, __bf16* __restrict__ d, int n)
{
    for (int i = blockIdx.x * blockDim.x + threadIdx.x; i < n;
         i += gridDim.x * blockDim.x)
        d[i] = (__bf16)s[i];
}

// W_out [V, 2H] -> bf16 padded to [WOUT_PAD, 2H] (zero rows beyond V)
__global__ void k_cvt_wout(const float* __restrict__ s, __bf16* __restrict__ d)
{
    const size_t total = (size_t)WOUT_PAD * D2H;
    for (size_t i = blockIdx.x * (size_t)blockDim.x + threadIdx.x; i < total;
         i += (size_t)gridDim.x * blockDim.x) {
        int r = (int)(i >> 10);
        int c = (int)(i & 1023);
        float v = (r < V_) ? s[(size_t)r * D2H + c] : 0.f;
        d[i] = (__bf16)v;
    }
}

__global__ void k_bias(const float* __restrict__ bih, const float* __restrict__ bhh,
                       float* __restrict__ bias)
{
    int i = blockIdx.x * blockDim.x + threadIdx.x;
    if (i < G4H) bias[i] = bih[i] + bhh[i];
}

__global__ void k_init_hc(const float* __restrict__ h0, const float* __restrict__ c0,
                          __bf16* __restrict__ hbf, float* __restrict__ ccur)
{
    int i = blockIdx.x * blockDim.x + threadIdx.x;
    if (i < B_ * H_) { hbf[i] = (__bf16)h0[i]; ccur[i] = c0[i]; }
}

// ---------------------------------------------------------------------------
// LSTM cell elementwise (one thread per (b, j)), torch gate order i,f,g,o.
// Sums the SPLITK_H hidden-GEMM partials, writes hy/cy to results[t], applies
// the skip residual (+= results[t-2]) to the carried state, and emits the
// carried h as bf16 for the next hidden GEMM.
// ---------------------------------------------------------------------------
__global__ void k_cell(const float* __restrict__ gatesX, const float* __restrict__ gatesH,
                       const float* __restrict__ bias, float* __restrict__ ccur,
                       __bf16* __restrict__ hbf, float* __restrict__ results,
                       int t, int do_skip)
{
    int i = blockIdx.x * blockDim.x + threadIdx.x;
    if (i >= B_ * H_) return;
    int b = i >> 9;
    int j = i & 511;
    const float* gx = gatesX + (size_t)(t * B_ + b) * G4H;
    float gi = gx[j]          + bias[j];
    float gf = gx[H_ + j]     + bias[H_ + j];
    float gg = gx[2 * H_ + j] + bias[2 * H_ + j];
    float go = gx[3 * H_ + j] + bias[3 * H_ + j];
    #pragma unroll
    for (int s = 0; s < SPLITK_H; ++s) {
        const float* gh = gatesH + (size_t)s * (B_ * G4H) + (size_t)b * G4H;
        gi += gh[j];
        gf += gh[H_ + j];
        gg += gh[2 * H_ + j];
        go += gh[3 * H_ + j];
    }
    float cx = ccur[i];
    float cy = sigmoidf_(gf) * cx + sigmoidf_(gi) * tanhf(gg);
    float hy = sigmoidf_(go) * tanhf(cy);
    size_t rrow = (size_t)(t * B_ + b) * D2H;
    results[rrow + j]      = hy;
    results[rrow + H_ + j] = cy;
    float hn = hy, cn = cy;
    if (do_skip) {
        size_t prow = (size_t)((t - 2) * B_ + b) * D2H;
        hn += results[prow + j];
        cn += results[prow + H_ + j];
    }
    ccur[i] = cn;
    hbf[i]  = (__bf16)hn;
}

// ---------------------------------------------------------------------------
// pack_padded_sequence indices: lengths sorted descending so
// cnt[t] = #(lengths > t); idx is time-major concat of rows 0..cnt[t)-1.
// ---------------------------------------------------------------------------
__global__ void k_indices(const int* __restrict__ lengths, int* __restrict__ idx)
{
    __shared__ int cnt[T_];
    __shared__ int off[T_];
    int tid = threadIdx.x;
    if (tid < T_) {
        int c = 0;
        for (int b = 0; b < B_; ++b) c += (lengths[b] > tid) ? 1 : 0;
        cnt[tid] = c;
    }
    __syncthreads();
    if (tid == 0) {
        int s = 0;
        for (int t = 0; t < T_; ++t) { off[t] = s; s += cnt[t]; }
    }
    __syncthreads();
    for (int t = 0; t < T_; ++t)
        if (tid < cnt[t]) idx[off[t] + tid] = t * B_ + tid;
}

__global__ void k_gather(const float* __restrict__ results, const int* __restrict__ idx,
                         __bf16* __restrict__ dst, int Ntot, int Npad)
{
    const size_t total = (size_t)Npad * D2H;
    for (size_t i = blockIdx.x * (size_t)blockDim.x + threadIdx.x; i < total;
         i += (size_t)gridDim.x * blockDim.x) {
        int n = (int)(i >> 10);
        int e = (int)(i & 1023);
        float v = 0.f;
        if (n < Ntot) v = results[(size_t)idx[n] * D2H + e];
        dst[i] = (__bf16)v;
    }
}

// ---------------------------------------------------------------------------
extern "C" void kernel_launch(void* const* d_in, const int* in_sizes, int n_in,
                              void* d_out, int out_size, void* d_ws, size_t ws_size,
                              hipStream_t stream)
{
    const float* img  = (const float*)d_in[0];
    const float* txt  = (const float*)d_in[1];
    const float* h0   = (const float*)d_in[2];
    const float* c0   = (const float*)d_in[3];
    const float* emb  = (const float*)d_in[4];
    const float* Wih  = (const float*)d_in[5];
    const float* Whh  = (const float*)d_in[6];
    const float* bih  = (const float*)d_in[7];
    const float* bhh  = (const float*)d_in[8];
    const float* Wout = (const float*)d_in[9];
    const float* bout = (const float*)d_in[10];
    const int*   answer  = (const int*)d_in[11];
    const int*   lengths = (const int*)d_in[12];
    float* out = (float*)d_out;

    const int Ntot = out_size / V_;             // sum(lengths)
    const int Npad = (Ntot + 127) & ~127;       // pad to 128-row block tiles

    // workspace carve (256B aligned regions)
    char* p = (char*)d_ws;
    auto carve = [&](size_t bytes) -> char* {
        char* r = p; p += (bytes + 255) & ~(size_t)255; return r;
    };
    __bf16* Xbf    = (__bf16*)carve((size_t)T_ * B_ * E_ * sizeof(__bf16));
    __bf16* Wihb   = (__bf16*)carve((size_t)G4H * E_ * sizeof(__bf16));
    __bf16* Whhb   = (__bf16*)carve((size_t)G4H * H_ * sizeof(__bf16));
    __bf16* Woutb  = (__bf16*)carve((size_t)WOUT_PAD * D2H * sizeof(__bf16));
    float*  bias   = (float*)carve((size_t)G4H * sizeof(float));
    float*  gatesX = (float*)carve((size_t)T_ * B_ * G4H * sizeof(float));
    float*  gatesH = (float*)carve((size_t)SPLITK_H * B_ * G4H * sizeof(float));
    float*  ccur   = (float*)carve((size_t)B_ * H_ * sizeof(float));
    __bf16* hbf    = (__bf16*)carve((size_t)B_ * H_ * sizeof(__bf16));
    float*  results= (float*)carve((size_t)T_ * B_ * D2H * sizeof(float));
    int*    idxbuf = (int*)carve((size_t)T_ * B_ * sizeof(int));
    __bf16* packedA= (__bf16*)carve((size_t)T_ * B_ * D2H * sizeof(__bf16));
    (void)ws_size; (void)n_in; (void)in_sizes;

    // --- stage 0: conversions / setup ---
    k_build_x<<<1024, 256, 0, stream>>>(img, txt, emb, answer, Xbf);
    k_cvt<<<1024, 256, 0, stream>>>(Wih, Wihb, G4H * E_);
    k_cvt<<<1024, 256, 0, stream>>>(Whh, Whhb, G4H * H_);
    k_cvt_wout<<<1024, 256, 0, stream>>>(Wout, Woutb);
    k_bias<<<8, 256, 0, stream>>>(bih, bhh, bias);
    k_init_hc<<<(B_ * H_ + 255) / 256, 256, 0, stream>>>(h0, c0, hbf, ccur);

    // --- stage 1: all-timestep input GEMM  gatesX = X * W_ih^T ---
    // 64x64 wave tiles (MT=NT=4): high arithmetic intensity, 1024 blocks.
    {
        dim3 g(G4H / 128, (T_ * B_) / 128, 1);          // (16, 64)
        k_gemm<4, 4><<<g, 128, 0, stream>>>(Xbf, Wihb, gatesX, E_, E_, G4H,
                                            T_ * B_, G4H, nullptr, 0);
    }

    // --- stage 2: sequential LSTM: split-K hidden GEMM + cell per step ---
    // MT=2,NT=4 tiles with 4-way deterministic split-K -> 256 blocks/step.
    for (int t = 0; t < T_; ++t) {
        dim3 gh(G4H / 128, B_ / 64, SPLITK_H);          // (16, 4, 4)
        k_gemm<2, 4><<<gh, 128, 0, stream>>>(hbf, Whhb, gatesH,
                                             H_ / SPLITK_H, H_, G4H,
                                             B_, G4H, nullptr,
                                             (size_t)B_ * G4H);
        int ds = (t >= 2 && (t % 3) == 0) ? 1 : 0;
        k_cell<<<(B_ * H_) / 256, 256, 0, stream>>>(gatesX, gatesH, bias, ccur,
                                                    hbf, results, t, ds);
    }

    // --- stage 3: pack + projection ---
    k_indices<<<1, 256, 0, stream>>>(lengths, idxbuf);
    k_gather<<<2048, 256, 0, stream>>>(results, idxbuf, packedA, Ntot, Npad);
    {
        dim3 go(WOUT_PAD / 128, Npad / 128, 1);         // (24, Npad/128)
        k_gemm<4, 4><<<go, 128, 0, stream>>>(packedA, Woutb, out, D2H, D2H, V_,
                                             Ntot, V_, bout, 0);
    }
}
